// LIFLayer_6124623364549
// MI455X (gfx1250) — compile-verified, matrix operand order checked by reading, not run on hardware
//
#include <hip/hip_runtime.h>
#include <hip/hip_bf16.h>

typedef __attribute__((ext_vector_type(16))) __bf16 v16bf;
typedef __attribute__((ext_vector_type(8)))  __bf16 v8bf;
typedef __attribute__((ext_vector_type(8)))  float  v8f;

#define T_LEN   512
#define IN_DIM  1024
#define OUT_DIM 1024
#define B_DIM   64
#define TT      256                     // t rows per block macro-tile (scan tile)
#define KC      128                     // K chunk staged in LDS per pass
#define ARPITCH (KC + 8)                // 136 bf16 = 272B row stride (16B aligned, 68 dw %64 = 4)
#define A_BYTES (TT * ARPITCH * 2)      // 69632
#define CPITCH  65
#define C_BYTES (TT * CPITCH * 4)       // 66560
#define SMEM_BYTES (A_BYTES + C_BYTES)  // ~136 KB dynamic LDS -> 2 blocks / 320KB WGP

// ---------------- W fp32 -> bf16 pre-convert (2 MB into workspace) ----------------
__global__ __launch_bounds__(256) void cvt_w_bf16(const float* __restrict__ w,
                                                  __bf16* __restrict__ o) {
    int i = (blockIdx.x * 256 + threadIdx.x) * 4;   // OUT*IN = 1M, divisible by 4
    float4 v = *(const float4*)(w + i);
    o[i + 0] = (__bf16)v.x;
    o[i + 1] = (__bf16)v.y;
    o[i + 2] = (__bf16)v.z;
    o[i + 3] = (__bf16)v.w;
}

// unaligned-safe 32B bf16 fragment load (two 16B halves)
__device__ inline v16bf load_frag16(const __bf16* p) {
    v8bf lo = *(const v8bf*)p;
    v8bf hi = *(const v8bf*)(p + 8);
    return __builtin_shufflevector(lo, hi, 0,1,2,3,4,5,6,7,8,9,10,11,12,13,14,15);
}

// ---------------- Fused GEMM (WMMA bf16, 4x4 register-blocked) + LIF scan ----------
// grid = (OUT_DIM/64, B_DIM); block = 128 (4 waves).
// Block macro-tile: 256t x 64n, K staged in 128-chunks.
// Wave w owns t rows [w*64, w*64+64) and all 64 n: 4 A-frags x 4 B-frags
// preloaded per k-step -> 16 dependency-free WMMAs per 8KB of operands.
__global__ __launch_bounds__(128) void lif_fused(const float*  __restrict__ x,
                                                 const __bf16* __restrict__ Wb,
                                                 const float*  __restrict__ bias,
                                                 float*        __restrict__ out) {
    extern __shared__ char smem[];
    __bf16 (*aTile)[ARPITCH] = (__bf16 (*)[ARPITCH])smem;           // [256][136] bf16
    float  (*cTile)[CPITCH]  = (float (*)[CPITCH])(smem + A_BYTES); // [256][65]  f32

    const int b     = blockIdx.y;
    const int n0    = blockIdx.x * 64;
    const int tid   = threadIdx.x;
    const int wave  = tid >> 5;
    const int lane  = tid & 31;
    const int lrow  = lane & 15;          // fragment row (A: t, B: n-col)
    const int khalf = (lane >> 4) * 16;   // K offset for upper half-wave
    const int mhi   = 8 * (lane >> 4);    // C-layout: lane holds M = j + mhi

    // B-matrix sources: W rows (n0 + f*16 + lrow), contiguous K (B = W^T)
    const __bf16* bbase[4];
    float bfrag[4];
    #pragma unroll
    for (int f = 0; f < 4; ++f) {
        bbase[f] = Wb + (size_t)(n0 + f * 16 + lrow) * IN_DIM + khalf;
        bfrag[f] = bias[n0 + f * 16 + lrow];   // lane's C column bias
    }

    float mem = 0.0f, spk = 0.0f;
    const float* xb = x + (size_t)b * T_LEN * IN_DIM;

    for (int T0 = 0; T0 < T_LEN; T0 += TT) {     // 2 macro-tiles
        // 16 independent accumulators, bias folded into C init (exact in f32)
        v8f acc[16];
        #pragma unroll
        for (int a = 0; a < 4; ++a)
            #pragma unroll
            for (int f = 0; f < 4; ++f)
                #pragma unroll
                for (int j = 0; j < 8; ++j)
                    acc[a * 4 + f][j] = bfrag[f];

        for (int kc = 0; kc < IN_DIM; kc += KC) {   // 8 K-chunks
            // ---- stage x[b, T0:T0+256, kc:kc+128] fp32 -> bf16 LDS ----
            const float* xt = xb + (size_t)T0 * IN_DIM + kc;
            #pragma unroll 4
            for (int i = 0; i < 64; ++i) {
                int flat = i * 512 + tid * 4;       // 4 rows x 512B contiguous per iter
                int r = flat >> 7;
                int c = flat & (KC - 1);
                float4 v = *(const float4*)(xt + (size_t)r * IN_DIM + c);
                __bf16* d = &aTile[r][c];
                d[0] = (__bf16)v.x; d[1] = (__bf16)v.y;
                d[2] = (__bf16)v.z; d[3] = (__bf16)v.w;
            }
            __syncthreads();

            // ---- 16 WMMAs per k-step, all operands preloaded ----
            #pragma unroll
            for (int k0 = 0; k0 < KC; k0 += 32) {
                v16bf bf[4];                         // global (L2-resident W) first
                #pragma unroll
                for (int f = 0; f < 4; ++f)
                    bf[f] = load_frag16(bbase[f] + kc + k0);
                v16bf af[4];                         // then LDS A-frags
                #pragma unroll
                for (int a = 0; a < 4; ++a)
                    af[a] = load_frag16(&aTile[wave * 64 + a * 16 + lrow][k0 + khalf]);
                #pragma unroll
                for (int a = 0; a < 4; ++a)
                    #pragma unroll
                    for (int f = 0; f < 4; ++f)
                        acc[a * 4 + f] = __builtin_amdgcn_wmma_f32_16x16x32_bf16(
                                false, af[a], false, bf[f], (short)0, acc[a * 4 + f],
                                false, false);
            }
            __syncthreads();   // aTile reads done before next chunk overwrites
        }

        // ---- spill C: lane holds M = j + mhi, N = lrow ----
        #pragma unroll
        for (int a = 0; a < 4; ++a)
            #pragma unroll
            for (int f = 0; f < 4; ++f)
                #pragma unroll
                for (int j = 0; j < 8; ++j)
                    cTile[wave * 64 + a * 16 + j + mhi][f * 16 + lrow] = acc[a * 4 + f][j];
        __syncthreads();

        // ---- sequential LIF scan over the 256 t-steps of this macro-tile ----
        if (tid < 64) {
            #pragma unroll 4
            for (int tt = 0; tt < TT; ++tt) {
                float inp = cTile[tt][tid];          // bias already included
                mem = 0.5f * (mem + inp);            // mem += (inp - mem)/TAU, TAU=2
                bool fire = (mem >= 1.0f);
                spk = fire ? 1.0f : 0.0f;
                mem = fire ? 0.0f : mem;
            }
        }
        __syncthreads();
    }

    if (tid < 64) {
        out[(size_t)b * OUT_DIM + n0 + tid] = mem;                               // mem_final
        out[(size_t)B_DIM * OUT_DIM + (size_t)b * OUT_DIM + n0 + tid] = spk;     // spk_final
    }
}

extern "C" void kernel_launch(void* const* d_in, const int* in_sizes, int n_in,
                              void* d_out, int out_size, void* d_ws, size_t ws_size,
                              hipStream_t stream) {
    const float* x    = (const float*)d_in[0];   // [64, 512, 1024]
    const float* W    = (const float*)d_in[1];   // [1024, 1024]
    const float* bias = (const float*)d_in[2];   // [1024]
    float* out        = (float*)d_out;           // mem[64,1024] ++ spk[64,1024]
    __bf16* Wb        = (__bf16*)d_ws;           // 2 MB bf16 copy of W

    // allow >64KB dynamic LDS for the fused kernel
    (void)hipFuncSetAttribute((const void*)lif_fused,
                              hipFuncAttributeMaxDynamicSharedMemorySize, SMEM_BYTES);

    // 1) W -> bf16 (1M elems, 4 per thread)
    cvt_w_bf16<<<(OUT_DIM * IN_DIM) / (256 * 4), 256, 0, stream>>>(W, Wb);

    // 2) fused GEMM + LIF scan
    dim3 grid(OUT_DIM / 64, B_DIM);
    lif_fused<<<grid, 128, SMEM_BYTES, stream>>>(x, Wb, bias, out);
}
